// KANLayer_30588757082465
// MI455X (gfx1250) — matrix operand order repaired and checked by reading, not compiled, not run on hardware
//
#include <hip/hip_runtime.h>
#include <stdint.h>

typedef __attribute__((ext_vector_type(16))) _Float16 v16h;
typedef __attribute__((ext_vector_type(8)))  float    v8f;

#define BATCH    1024
#define IN_DIM   256
#define OUT_DIM  256
#define NKNOT    19            // spline intervals = coeffs.shape[2]
#define KPAD     24            // knots padded to 3 groups of 8
#define KDIM     (KPAD * 4)    // 96 contraction elements per input dim
#define XMINF    (-5.0f)
#define HF       (10.0f / 19.0f)
#define INVHF    (19.0f / 10.0f)

// LDS staging: 64 output rows of 96 halves (192B) + 16B pad -> conflict-free b128 reads
#define BROW_BYTES  (KDIM * 2)        // 192
#define BROW_STRIDE (BROW_BYTES + 16) // 208
#define SLAB_BYTES  (64 * BROW_STRIDE)

__device__ __forceinline__ uint32_t pack2h(float lo, float hi) {
    union { _Float16 h[2]; uint32_t u; } p;
    p.h[0] = (_Float16)lo;
    p.h[1] = (_Float16)hi;
    return p.u;
}

// CDNA5 async global->LDS copy (ASYNCcnt-tracked, bypasses VGPRs). GV mode.
__device__ __forceinline__ void async_g2l_b128(uint32_t lds_byte_addr, const void* gaddr) {
    asm volatile("global_load_async_to_lds_b128 %0, %1, off"
                 :: "v"(lds_byte_addr), "v"(gaddr) : "memory");
}
__device__ __forceinline__ void wait_async0() {
    asm volatile("s_wait_asynccnt 0x0" ::: "memory");
}

// LDS byte offset = low 32 bits of the flat address (aperture truncation).
__device__ __forceinline__ uint32_t lds_off(const void* p) {
    return (uint32_t)(uintptr_t)p;
}

// ---------------------------------------------------------------------------
// Kernel 1: repack coeffs (O, I, 19, 4) f32  ->  Bt f16 [i][o][K=96]
// K index = knot*4 + power; knots 19..23 are zero pad (idx<=18 never matches).
// ---------------------------------------------------------------------------
__global__ void kan_pack_coeffs(const float* __restrict__ coeffs,
                                _Float16* __restrict__ Bt) {
    const int o = blockIdx.x;    // 0..255
    const int i = threadIdx.x;   // 0..255
    const float*   src = coeffs + ((size_t)o * IN_DIM + i) * (NKNOT * 4);
    _Float16*      dst = Bt + ((size_t)i * OUT_DIM + o) * KDIM;
    #pragma unroll 4
    for (int e = 0; e < NKNOT * 4; ++e) dst[e] = (_Float16)src[e];
    #pragma unroll
    for (int e = NKNOT * 4; e < KDIM; ++e) dst[e] = (_Float16)0.0f;
}

// Stage one 64-row x 192B B slab into an LDS buffer: 3 async b128 per thread.
__device__ __forceinline__ void stage_slab_async(const _Float16* __restrict__ brow,
                                                 unsigned char* buf, int tid) {
    #pragma unroll
    for (int s = 0; s < 3; ++s) {
        const int linear = tid + s * 256;   // 0..767 = 64 rows * 12 x b128
        const int orow   = linear / 12;
        const int seg    = linear % 12;
        async_g2l_b128(lds_off(buf + orow * BROW_STRIDE + seg * 16),
                       (const void*)(brow + (size_t)orow * KDIM + seg * 8));
    }
}

// ---------------------------------------------------------------------------
// Kernel 2: fused spline-basis construction + WMMA GEMM
//   grid  = (BATCH/64, OUT_DIM/64), block = 256 (8 waves, wave32)
//   wave (wb, wo): A rows = 16 batch rows, C tiles = two 16x16 out tiles
//   Double-buffered LDS; slab i+1 streams in via async-to-LDS under slab-i WMMAs.
// ---------------------------------------------------------------------------
__launch_bounds__(256)
__global__ void kan_wmma_kernel(const float* __restrict__ x,
                                const _Float16* __restrict__ Bt,
                                const float* __restrict__ bias,
                                float* __restrict__ out) {
    __shared__ unsigned char smem[2][SLAB_BYTES];   // 2 x 13 KB

    const int tid  = threadIdx.x;
    const int lane = tid & 31;
    const int wv   = tid >> 5;        // 0..7
    const int wb   = wv & 3;          // batch sub-tile 0..3
    const int wo   = wv >> 2;         // 0..1 (32-wide out strip)
    const int hl   = (lane >> 4) & 1; // lane half (A: K+8, B: K+8, C: M+8)
    const int l16  = lane & 15;

    const int bbase = blockIdx.x * 64;
    const int obase = blockIdx.y * 64;
    const int row   = bbase + wb * 16 + l16;   // this lane's batch row (A M index)

    v8f c0 = {};
    v8f c1 = {};

    // Prologue: stage slab for i=0 into buffer 0.
    stage_slab_async(Bt + ((size_t)0 * OUT_DIM + obase) * KDIM, &smem[0][0], tid);
    wait_async0();
    __syncthreads();

    for (int i = 0; i < IN_DIM; ++i) {
        unsigned char* cur = &smem[i & 1][0];

        // Kick off next slab into the other buffer (overlaps with WMMAs below).
        if (i + 1 < IN_DIM)
            stage_slab_async(Bt + ((size_t)(i + 1) * OUT_DIM + obase) * KDIM,
                             &smem[(i + 1) & 1][0], tid);

        // ---- per-lane spline basis: idx, t, powers (packed f16 pairs) ----
        const float xv = x[(size_t)row * IN_DIM + i];
        int idx = (int)floorf((xv - XMINF) * INVHF);
        idx = idx < 0 ? 0 : (idx > NKNOT - 1 ? NKNOT - 1 : idx);
        const float t  = xv - (XMINF + (float)idx * HF);
        const float t2 = t * t;
        const float t3 = t2 * t;
        const uint32_t u01 = pack2h(1.0f, t);   // powers p=0,1
        const uint32_t u23 = pack2h(t2, t3);    // powers p=2,3

        #pragma unroll
        for (int kg = 0; kg < 3; ++kg) {        // 8 knots per K=32 chunk
            // ---- A fragment (16x32 f16): VGPR v holds K = kc+8*hl, kc+8*hl+1 ----
            union { uint32_t u[8]; v16h h; } af;
            #pragma unroll
            for (int v = 0; v < 8; ++v) {
                const int kc = (v < 4) ? (2 * v) : (16 + 2 * (v - 4));
                const int kk = (kc >> 2) + (hl << 1);     // knot within group
                const uint32_t pk = ((kc & 2) == 0) ? u01 : u23;
                af.u[v] = (idx == kg * 8 + kk) ? pk : 0u;
            }
            // ---- B fragments (32x16 f16) from LDS: lane n=l16, K=8*hl+e (e<8), 16+8*hl+(e-8) ----
            union { uint4 q[2]; v16h h; } bf0, bf1;
            {
                const unsigned char* p =
                    cur + (wo * 32 + l16) * BROW_STRIDE + kg * 64 + hl * 16;
                bf0.q[0] = *(const uint4*)(p);
                bf0.q[1] = *(const uint4*)(p + 32);
                const unsigned char* q = p + 16 * BROW_STRIDE;
                bf1.q[0] = *(const uint4*)(q);
                bf1.q[1] = *(const uint4*)(q + 32);
            }
            c0 = __builtin_amdgcn_wmma_f32_16x16x32_f16(
                     false, af.h, false, bf0.h, (short)0, c0, false, false);
            c1 = __builtin_amdgcn_wmma_f32_16x16x32_f16(
                     false, af.h, false, bf1.h, (short)0, c1, false, false);
        }

        // Publish next slab: my async writes landed, then block-wide barrier.
        wait_async0();
        __syncthreads();
    }

    // ---- epilogue: C layout M = r + 8*hl, N = l16; add bias; store f32 ----
    const int ocol = obase + wo * 32 + l16;
    const float bv0 = bias[ocol];
    const float bv1 = bias[ocol + 16];
    #pragma unroll
    for (int r = 0; r < 8; ++r) {
        const int brow = bbase + wb * 16 + r + hl * 8;
        float* op = out + (size_t)brow * OUT_DIM + ocol;
        op[0]  = c0[r] + bv0;
        op[16] = c1[r] + bv1;
    }
}

// ---------------------------------------------------------------------------
// Launch: d_in = { x (1024x256 f32), coeffs (256x256x19x4 f32), bias (256 f32) }
// d_out = 1024x256 f32. d_ws holds the 12.6 MB f16 repacked coeff table.
// ---------------------------------------------------------------------------
extern "C" void kernel_launch(void* const* d_in, const int* in_sizes, int n_in,
                              void* d_out, int out_size, void* d_ws, size_t ws_size,
                              hipStream_t stream) {
    const float* x      = (const float*)d_in[0];
    const float* coeffs = (const float*)d_in[1];
    const float* bias   = (const float*)d_in[2];
    float*       out    = (float*)d_out;
    _Float16*    Bt     = (_Float16*)d_ws;   // needs IN_DIM*OUT_DIM*KDIM*2 = 12.6 MB

    kan_pack_coeffs<<<dim3(OUT_DIM), dim3(IN_DIM), 0, stream>>>(coeffs, Bt);

    dim3 grid(BATCH / 64, OUT_DIM / 64);
    kan_wmma_kernel<<<grid, dim3(256), 0, stream>>>(x, Bt, bias, out);
}
